// MoE_77352361001110
// MI455X (gfx1250) — matrix operand order repaired and checked by reading, not compiled, not run on hardware
//
#include <hip/hip_runtime.h>
#include <hip/hip_bf16.h>
#include <math.h>

#define EMB 1024
#define HID 1536
#define NE  8
#define TOK 8192            // B*N = 4*2048

typedef __attribute__((ext_vector_type(16))) __bf16 v16bf;
typedef __attribute__((ext_vector_type(8)))  float  v8f;

#define XPITCH 1032         // 1024 + 8 bf16 pad (16B-aligned rows, bank-shifted)
#define HPITCH 136          // 128 + 8

#define FRAG_ELEMS 512      // one 32x16 bf16 B-fragment = 512 elems = 1KB

__device__ __forceinline__ __bf16 f2bf(float f) { return (__bf16)f; }

// A fragment (16x32 bf16) from LDS; element s of lane half h is k = 16h + s (contiguous order).
__device__ __forceinline__ v16bf load_a_lds(const __bf16* base, int pitch,
                                            int rowbase, int kbase) {
  int lane = threadIdx.x & 31;
  int row  = rowbase + (lane & 15);
  int kh   = (lane >> 4) << 4;                // 0 or 16
  const __bf16* p = base + row * pitch + kbase + kh;
  v16bf r;
#pragma unroll
  for (int i = 0; i < 8; ++i) { r[i] = p[i]; r[8 + i] = p[8 + i]; }
  return r;
}

// Slow-path B fragment (32x16) gathered from a global f32 row-major matrix.
__device__ __forceinline__ v16bf load_b_glb(const float* __restrict__ B, int ld,
                                            int kbase, int nbase) {
  int lane = threadIdx.x & 31;
  int n  = nbase + (lane & 15);
  int k0 = kbase + ((lane >> 4) << 4);
  const float* p = B + (size_t)k0 * ld + n;
  v16bf r;
#pragma unroll
  for (int i = 0; i < 16; ++i) r[i] = f2bf(p[(size_t)i * ld]);
  return r;
}

// Fast-path B fragment: pre-packed bf16, lane-major 32B per lane, 1KB per fragment.
__device__ __forceinline__ v16bf load_b_packed(const __bf16* __restrict__ Wp,
                                               int frag) {
  int lane = threadIdx.x & 31;
  return *(const v16bf*)(Wp + (size_t)frag * FRAG_ELEMS + lane * 16);
}

// ---------------- Weight packer: f32 [NE][K][N] -> bf16 fragments.
// nt_outer=1: frag order [e][n_tile][k_tile] (GEMM1/W1);
// nt_outer=0: frag order [e][k_tile][n_tile] (GEMM2/W2).
__global__ void pack_w_kernel(const float* __restrict__ W,
                              __bf16* __restrict__ Wp,
                              int K, int N, int nt_outer) {
  const int gid  = blockIdx.x * 256 + threadIdx.x;
  const int lane = gid & 31;
  const int frag = gid >> 5;
  const int kt_n = K / 32, nt_n = N / 16, fpe = kt_n * nt_n;
  const int e  = frag / fpe;
  const int fi = frag - e * fpe;
  int k_tile, n_tile;
  if (nt_outer) { n_tile = fi / kt_n; k_tile = fi - n_tile * kt_n; }
  else          { k_tile = fi / nt_n; n_tile = fi - k_tile * nt_n; }
  const int n  = n_tile * 16 + (lane & 15);
  const int k0 = k_tile * 32 + ((lane >> 4) << 4);
  const float* p = W + (size_t)e * K * N + (size_t)k0 * N + n;
  v16bf r;
#pragma unroll
  for (int i = 0; i < 16; ++i) r[i] = f2bf(p[(size_t)i * N]);
  *(v16bf*)(Wp + (size_t)frag * FRAG_ELEMS + lane * 16) = r;
}

// ---------------- Router: logits = x@Wr + br, top-k -> scale[t][e] in {0, 1/k}
__global__ void router_kernel(const float* __restrict__ x,
                              const float* __restrict__ Wr,
                              const float* __restrict__ br,
                              const int* __restrict__ kp,
                              float* __restrict__ scale) {
  int wave = threadIdx.x >> 5;
  int lane = threadIdx.x & 31;
  int t = blockIdx.x * 8 + wave;
  const float* xr = x + (size_t)t * EMB;
  float acc[NE];
#pragma unroll
  for (int e = 0; e < NE; ++e) acc[e] = 0.f;
  for (int j = 0; j < EMB / 32; ++j) {
    int row = j * 32 + lane;
    float xv = xr[row];
    const float* wr = Wr + (size_t)row * NE;
#pragma unroll
    for (int e = 0; e < NE; ++e) acc[e] = fmaf(xv, wr[e], acc[e]);
  }
#pragma unroll
  for (int off = 16; off > 0; off >>= 1) {
#pragma unroll
    for (int e = 0; e < NE; ++e) acc[e] += __shfl_xor(acc[e], off, 32);
  }
  if (lane == 0) {
    int kk = *kp;
    float lg[NE], sc[NE];
#pragma unroll
    for (int e = 0; e < NE; ++e) { lg[e] = acc[e] + br[e]; sc[e] = 0.f; }
    float inv = 1.0f / (float)kk;
    for (int s = 0; s < kk; ++s) {
      int am = 0; float bv = lg[0];
#pragma unroll
      for (int e = 1; e < NE; ++e) { if (lg[e] > bv) { bv = lg[e]; am = e; } }
      sc[am] = inv; lg[am] = -INFINITY;
    }
#pragma unroll
    for (int e = 0; e < NE; ++e) scale[(size_t)t * NE + e] = sc[e];
  }
}

// ---------------- Fused dense MoE FFN with bf16 WMMA.
// Block = 8 waves, M = 32 tokens. h never leaves LDS; routing scale folded into h.
// B fragments use a 2-deep software pipeline (two live buffers) in both GEMMs.
template <bool PACKED>
__global__ void __launch_bounds__(256)
moe_fused_kernel(const float* __restrict__ x,
                 const float* __restrict__ W1,
                 const float* __restrict__ b1,
                 const float* __restrict__ W2,
                 const float* __restrict__ b2,
                 const __bf16* __restrict__ W1p,
                 const __bf16* __restrict__ W2p,
                 const float* __restrict__ scale,
                 float* __restrict__ out) {
  __shared__ float  xstage[16 * EMB];        // 64KB async-copy staging (reused 2x)
  __shared__ __bf16 x_lds[32 * XPITCH];      // 66KB resident bf16 x tile
  __shared__ __bf16 h_lds[32 * HPITCH];      // 8.7KB
  __shared__ float  s_lds[32 * NE];          // 1KB

  const int tid  = threadIdx.x;
  const int lane = tid & 31;
  const int w    = tid >> 5;
  const int t0   = blockIdx.x * 32;

  // Phase 0: stage x tile via async global->LDS DMA (f32), then convert to bf16.
  if (tid < 32 * NE) s_lds[tid] = scale[(size_t)t0 * NE + tid];
  for (int half = 0; half < 2; ++half) {
    for (int i = tid; i < 16 * (EMB / 4); i += 256) {
      int row = i >> 8;
      int c4  = i & 255;
      const float* g = x + (size_t)(t0 + half * 16 + row) * EMB + c4 * 4;
      unsigned ldsoff = (unsigned)(size_t)(&xstage[row * EMB + c4 * 4]);
      asm volatile("global_load_async_to_lds_b128 %0, %1, off"
                   :: "v"(ldsoff), "v"(g) : "memory");
    }
    asm volatile("s_wait_asynccnt 0x0" ::: "memory");
    __syncthreads();
    for (int i = tid; i < 16 * (EMB / 4); i += 256) {
      int row = i >> 8;
      int c4  = i & 255;
      const float4 v = ((const float4*)(&xstage[row * EMB]))[c4];
      __bf16* d = x_lds + (half * 16 + row) * XPITCH + c4 * 4;
      d[0] = f2bf(v.x); d[1] = f2bf(v.y); d[2] = f2bf(v.z); d[3] = f2bf(v.w);
    }
    __syncthreads();
  }

  v8f acc[2][8];
#pragma unroll
  for (int mt = 0; mt < 2; ++mt)
#pragma unroll
    for (int nt = 0; nt < 8; ++nt) { v8f z = {}; acc[mt][nt] = z; }

  for (int e = 0; e < NE; ++e) {
    const float*  W1e  = W1 + (size_t)e * EMB * HID;
    const float*  W2e  = W2 + (size_t)e * HID * EMB;
    const __bf16* W1pe = W1p + (size_t)e * (EMB / 32) * (HID / 16) * FRAG_ELEMS;
    const __bf16* W2pe = W2p + (size_t)e * (HID / 32) * (EMB / 16) * FRAG_ELEMS;
    for (int sb = 0; sb < HID / 128; ++sb) {
      const int hcol = sb * 128 + w * 16;   // this wave's global H-column base
      const int nt1  = sb * 8 + w;          // GEMM1 n_tile index
      if (PACKED)
        __builtin_prefetch(W2pe + (size_t)(sb * 4) * (EMB / 16) * FRAG_ELEMS, 0, 1);
      else
        __builtin_prefetch(W2e + (size_t)(sb * 128) * EMB + w * 128, 0, 1);

      // ---- GEMM1: h[:, hcol..hcol+16) = x_tile @ W1e (2-deep B pipeline) ----
      v8f h0 = {}, h1 = {};
      v16bf b0, b1v16;
      if constexpr (PACKED) {
        b0    = load_b_packed(W1pe, nt1 * (EMB / 32) + 0);
        b1v16 = load_b_packed(W1pe, nt1 * (EMB / 32) + 1);
      } else {
        b0    = load_b_glb(W1e, HID, 0,  hcol);
        b1v16 = load_b_glb(W1e, HID, 32, hcol);
      }
#pragma unroll 2
      for (int kk = 0; kk < EMB / 32; kk += 2) {
        v16bf a00 = load_a_lds(x_lds, XPITCH, 0,  kk * 32);
        v16bf a01 = load_a_lds(x_lds, XPITCH, 16, kk * 32);
        v16bf a10 = load_a_lds(x_lds, XPITCH, 0,  (kk + 1) * 32);
        v16bf a11 = load_a_lds(x_lds, XPITCH, 16, (kk + 1) * 32);
        v16bf p0 = b0, p1 = b1v16;          // snapshot before overwrite
        if (kk + 2 < EMB / 32) {
          if constexpr (PACKED) {
            b0    = load_b_packed(W1pe, nt1 * (EMB / 32) + kk + 2);
            b1v16 = load_b_packed(W1pe, nt1 * (EMB / 32) + kk + 3);
          } else {
            b0    = load_b_glb(W1e, HID, (kk + 2) * 32, hcol);
            b1v16 = load_b_glb(W1e, HID, (kk + 3) * 32, hcol);
          }
        }
        h0 = __builtin_amdgcn_wmma_f32_16x16x32_bf16(false, a00, false, p0,
                                                     (short)0, h0, false, false);
        h1 = __builtin_amdgcn_wmma_f32_16x16x32_bf16(false, a01, false, p0,
                                                     (short)0, h1, false, false);
        h0 = __builtin_amdgcn_wmma_f32_16x16x32_bf16(false, a10, false, p1,
                                                     (short)0, h0, false, false);
        h1 = __builtin_amdgcn_wmma_f32_16x16x32_bf16(false, a11, false, p1,
                                                     (short)0, h1, false, false);
      }
      // bias + exact erf-GELU + routing scale, store bf16 h to LDS
      const float b1bias = b1[(size_t)e * HID + hcol + (lane & 15)];
      const int   mhi = (lane >> 4) << 3;
      const int   ncol = w * 16 + (lane & 15);
#pragma unroll
      for (int mt = 0; mt < 2; ++mt) {
        v8f cc = mt ? h1 : h0;
#pragma unroll
        for (int i = 0; i < 8; ++i) {
          int m = mt * 16 + i + mhi;
          float v = cc[i] + b1bias;
          float g = 0.5f * v * (1.0f + erff(v * 0.70710678118654752f));
          h_lds[m * HPITCH + ncol] = f2bf(s_lds[m * NE + e] * g);
        }
      }
      __syncthreads();

      // ---- GEMM2: acc += (scale*h) @ W2e (2-deep B pipeline) ----
#pragma unroll
      for (int kk2 = 0; kk2 < 4; ++kk2) {
        v16bf aH0 = load_a_lds(h_lds, HPITCH, 0,  kk2 * 32);
        v16bf aH1 = load_a_lds(h_lds, HPITCH, 16, kk2 * 32);
        const int fbase = (sb * 4 + kk2) * (EMB / 16) + w * 8;
        const int nbase2 = w * 128;
        const int kbase2 = sb * 128 + kk2 * 32;
        v16bf c0, c1;
        if constexpr (PACKED) {
          c0 = load_b_packed(W2pe, fbase + 0);
          c1 = load_b_packed(W2pe, fbase + 1);
        } else {
          c0 = load_b_glb(W2e, EMB, kbase2, nbase2 + 0);
          c1 = load_b_glb(W2e, EMB, kbase2, nbase2 + 16);
        }
#pragma unroll
        for (int nt = 0; nt < 8; nt += 2) {
          v16bf p0 = c0, p1 = c1;           // snapshot before overwrite
          if (nt + 2 < 8) {
            if constexpr (PACKED) {
              c0 = load_b_packed(W2pe, fbase + nt + 2);
              c1 = load_b_packed(W2pe, fbase + nt + 3);
            } else {
              c0 = load_b_glb(W2e, EMB, kbase2, nbase2 + (nt + 2) * 16);
              c1 = load_b_glb(W2e, EMB, kbase2, nbase2 + (nt + 3) * 16);
            }
          }
          acc[0][nt]     = __builtin_amdgcn_wmma_f32_16x16x32_bf16(false, aH0, false, p0,
                                              (short)0, acc[0][nt],     false, false);
          acc[1][nt]     = __builtin_amdgcn_wmma_f32_16x16x32_bf16(false, aH1, false, p0,
                                              (short)0, acc[1][nt],     false, false);
          acc[0][nt + 1] = __builtin_amdgcn_wmma_f32_16x16x32_bf16(false, aH0, false, p1,
                                              (short)0, acc[0][nt + 1], false, false);
          acc[1][nt + 1] = __builtin_amdgcn_wmma_f32_16x16x32_bf16(false, aH1, false, p1,
                                              (short)0, acc[1][nt + 1], false, false);
        }
      }
      __syncthreads();
    }
  }

  // Epilogue: out = acc + sum_e scale[t,e] * b2[e]  (full overwrite, no pre-zero)
  const int mhi = (lane >> 4) << 3;
#pragma unroll
  for (int nt = 0; nt < 8; ++nt) {
    int gcol = w * 128 + nt * 16 + (lane & 15);
    float b2v[NE];
#pragma unroll
    for (int e = 0; e < NE; ++e) b2v[e] = b2[(size_t)e * EMB + gcol];
#pragma unroll
    for (int mt = 0; mt < 2; ++mt)
#pragma unroll
      for (int i = 0; i < 8; ++i) {
        int m = mt * 16 + i + mhi;
        float bias = 0.f;
#pragma unroll
        for (int e = 0; e < NE; ++e) bias = fmaf(s_lds[m * NE + e], b2v[e], bias);
        out[(size_t)(t0 + m) * EMB + gcol] = acc[mt][nt][i] + bias;
      }
  }
}

extern "C" void kernel_launch(void* const* d_in, const int* in_sizes, int n_in,
                              void* d_out, int out_size, void* d_ws, size_t ws_size,
                              hipStream_t stream) {
  const float* x  = (const float*)d_in[0];
  const float* Wr = (const float*)d_in[1];
  const float* br = (const float*)d_in[2];
  const float* W1 = (const float*)d_in[3];
  const float* b1 = (const float*)d_in[4];
  const float* W2 = (const float*)d_in[5];
  const float* b2 = (const float*)d_in[6];
  const int*   kp = (const int*)d_in[7];
  float* out = (float*)d_out;

  const size_t scale_bytes = (size_t)TOK * NE * sizeof(float);          // 256 KB
  const size_t w_bytes     = (size_t)NE * EMB * HID * sizeof(__bf16);   // 24 MB each
  float*  scale = (float*)d_ws;
  __bf16* W1p   = (__bf16*)((char*)d_ws + scale_bytes);
  __bf16* W2p   = (__bf16*)((char*)d_ws + scale_bytes + w_bytes);
  const bool packed = ws_size >= scale_bytes + 2 * w_bytes;

  router_kernel<<<TOK / 8, 256, 0, stream>>>(x, Wr, br, kp, scale);
  if (packed) {
    const int nfrag = NE * (EMB / 32) * (HID / 16);   // same count for both
    pack_w_kernel<<<nfrag * 32 / 256, 256, 0, stream>>>(W1, W1p, EMB, HID, 1);
    pack_w_kernel<<<nfrag * 32 / 256, 256, 0, stream>>>(W2, W2p, HID, EMB, 0);
    moe_fused_kernel<true><<<TOK / 32, 256, 0, stream>>>(x, W1, b1, W2, b2,
                                                         W1p, W2p, scale, out);
  } else {
    moe_fused_kernel<false><<<TOK / 32, 256, 0, stream>>>(x, W1, b1, W2, b2,
                                                          W1p, W2p, scale, out);
  }
}
